// MatryoshkaSampledSoftmaxLoss_65807488909922
// MI455X (gfx1250) — compile-verified, matrix operand order checked by reading, not run on hardware
//
#include <hip/hip_runtime.h>
#include <hip/hip_bf16.h>

// ---------------------------------------------------------------------------
// Matryoshka sampled softmax loss for MI455X (gfx1250, wave32, WMMA).
//   h: (2048,1024) f32, W: (262144,1024) f32, tgt: (2048,) i32 -> scalar f32
// ---------------------------------------------------------------------------

#define VOCAB   262144
#define NROWS   2048
#define DIM     1024
#define CHUNKSZ 32
#define NCHUNK  64          // 2048/32
#define KCAND   2048        // candidates per chunk
#define SEGLEN  1024        // VOCAB / 256 segments

typedef __attribute__((ext_vector_type(16))) __bf16 v16bf;
typedef __attribute__((ext_vector_type(8)))  float  v8f;

// Build a 16x32(K) bf16 WMMA fragment for one lane from a row-major f32 row.
// CDNA5 16-bit A/B layout: lanes 0-15 hold K[0..7],K[16..23]; lanes 16-31 hold
// K[8..15],K[24..31] of the 32-wide K window -> two contiguous 32B chunks at
// (half*8) and (16+half*8).  rowptr already points at (row, k_base).
__device__ __forceinline__ v16bf frag_f32(const float* __restrict__ rowptr, int half) {
    const float4* p0 = (const float4*)(rowptr + half * 8);
    const float4* p1 = (const float4*)(rowptr + 16 + half * 8);
    float4 a = p0[0], b = p0[1], c = p1[0], d = p1[1];
    v16bf f;
    f[0]  = (__bf16)a.x; f[1]  = (__bf16)a.y; f[2]  = (__bf16)a.z; f[3]  = (__bf16)a.w;
    f[4]  = (__bf16)b.x; f[5]  = (__bf16)b.y; f[6]  = (__bf16)b.z; f[7]  = (__bf16)b.w;
    f[8]  = (__bf16)c.x; f[9]  = (__bf16)c.y; f[10] = (__bf16)c.z; f[11] = (__bf16)c.w;
    f[12] = (__bf16)d.x; f[13] = (__bf16)d.y; f[14] = (__bf16)d.z; f[15] = (__bf16)d.w;
    return f;
}

__device__ __forceinline__ v8f wmma_bf16(v16bf a, v16bf b, v8f c) {
    return __builtin_amdgcn_wmma_f32_16x16x32_bf16(false, a, false, b, (short)0, c,
                                                   false, false);
}

// online logsumexp helpers --------------------------------------------------
__device__ __forceinline__ void lse_accum(float& m, float& s, float x) {
    if (x > m) { s = s * __expf(m - x) + 1.0f; m = x; }
    else       { s += __expf(x - m); }
}
__device__ __forceinline__ void lse_combine(float& m, float& s, float m2, float s2) {
    float M = fmaxf(m, m2);
    s = s * __expf(m - M) + s2 * __expf(m2 - M);
    m = M;
}

__device__ __forceinline__ float block_sum(float v, float* red, int tid) {
    red[tid] = v; __syncthreads();
    for (int off = 128; off > 0; off >>= 1) {
        if (tid < off) red[tid] += red[tid + off];
        __syncthreads();
    }
    v = red[0]; __syncthreads();
    return v;
}

// ---------------------------------------------------------------------------
// 0) init: zero accumulators (buffers are poisoned by the harness)
// ---------------------------------------------------------------------------
__global__ void mat_init_kernel(float* wsn, float* out) {
    if (threadIdx.x == 0 && blockIdx.x == 0) {
        wsn[0] = 0.0f; wsn[1] = 0.0f; out[0] = 0.0f;
    }
}

// ---------------------------------------------------------------------------
// 1) sum of squares of W (full and first-64 dims), grid-stride float4
// ---------------------------------------------------------------------------
__global__ void mat_wnorm_kernel(const float* __restrict__ W, float* wsn) {
    __shared__ float r0[256];
    __shared__ float r1[256];
    const int tid = threadIdx.x;
    const size_t n4 = (size_t)VOCAB * DIM / 4;
    float s = 0.0f, sl = 0.0f;
    for (size_t i = (size_t)blockIdx.x * blockDim.x + tid; i < n4;
         i += (size_t)gridDim.x * blockDim.x) {
        float4 v = ((const float4*)W)[i];
        float d = v.x * v.x + v.y * v.y + v.z * v.z + v.w * v.w;
        s += d;
        // element index = 4*i, dim = (4*i)%1024 < 64  <=>  (i%256) < 16
        if ((i & 255) < 16) sl += d;
    }
    s  = block_sum(s,  r0, tid);
    sl = block_sum(sl, r1, tid);
    if (tid == 0) {
        atomicAdd(&wsn[0], s);
        atomicAdd(&wsn[1], sl);
    }
}

// ---------------------------------------------------------------------------
// 2) scout scan: scan_logits = scouts(512x64) x w_low(Vx64)^T via bf16 WMMA,
//    segmented argmax (one candidate per scout per 1024-row vocab segment).
//    grid = 256 segments * 4 scout-blocks, block = 256 (8 waves).
//    Two column tiles per inner iteration -> 2 independent WMMA chains.
// ---------------------------------------------------------------------------
__global__ void mat_scan_kernel(const float* __restrict__ h,
                                const float* __restrict__ W,
                                int* __restrict__ topidx) {
    const int bid  = blockIdx.x;
    const int seg  = bid & 255;
    const int sblk = bid >> 8;                 // 0..3 : block of 128 scouts
    const int tid  = threadIdx.x;
    const int lane = tid & 31, wave = tid >> 5;
    const int half = lane >> 4, l16 = lane & 15;

    // A fragments: 16 scouts (rows) x K=64, loaded once
    const int sg = sblk * 128 + wave * 16 + l16;     // global scout id 0..511
    const int c  = sg >> 3, s = sg & 7;
    const float* arow = h + (size_t)(c * CHUNKSZ + s * 4) * DIM;   // hc[c, s*4, :]
    v16bf a0 = frag_f32(arow + 0,  half);
    v16bf a1 = frag_f32(arow + 32, half);

    float vmax[8];
    int   vidx[8];
#pragma unroll
    for (int r = 0; r < 8; ++r) { vmax[r] = -3.0e38f; vidx[r] = 0; }

    for (int sub = 0; sub < 4; ++sub) {
        for (int ct = 0; ct < 16; ct += 2) {
            const int v0 = seg * SEGLEN + sub * 256 + ct * 16 + l16;
            const int v1 = v0 + 16;
            const float* brow0 = W + (size_t)v0 * DIM;   // w_low rows (K-major)
            const float* brow1 = W + (size_t)v1 * DIM;
            v16bf b00 = frag_f32(brow0 + 0,  half);
            v16bf b01 = frag_f32(brow0 + 32, half);
            v16bf b10 = frag_f32(brow1 + 0,  half);
            v16bf b11 = frag_f32(brow1 + 32, half);
            v8f acc0 = {}, acc1 = {};
            acc0 = wmma_bf16(a0, b00, acc0);
            acc1 = wmma_bf16(a0, b10, acc1);
            acc0 = wmma_bf16(a1, b01, acc0);
            acc1 = wmma_bf16(a1, b11, acc1);
#pragma unroll
            for (int r = 0; r < 8; ++r) {
                float x0 = acc0[r];        // (M=r+8*half, N=l16) -> vocab row v0
                float x1 = acc1[r];
                if (x0 > vmax[r]) { vmax[r] = x0; vidx[r] = v0; }
                if (x1 > vmax[r]) { vmax[r] = x1; vidx[r] = v1; }
            }
        }
    }
    // argmax across the 16 lanes that share a row group (halves are disjoint rows)
#pragma unroll
    for (int msk = 8; msk >= 1; msk >>= 1) {
#pragma unroll
        for (int r = 0; r < 8; ++r) {
            float ov = __shfl_xor(vmax[r], msk, 32);
            int   oi = __shfl_xor(vidx[r], msk, 32);
            if (ov > vmax[r]) { vmax[r] = ov; vidx[r] = oi; }
        }
    }
    if (l16 == 0) {
#pragma unroll
        for (int r = 0; r < 8; ++r) {
            const int M   = r + 8 * half;
            const int sg2 = sblk * 128 + wave * 16 + M;
            const int c2  = sg2 >> 3, s2 = sg2 & 7;
            topidx[c2 * KCAND + s2 * 256 + seg] = vidx[r];
        }
    }
}

// ---------------------------------------------------------------------------
// 3) candidate logits: neg_m = hc(32x1024) x Wcand^T and neg_a (first 64 dims)
//    grid = 64 chunks * 8 candidate-blocks, block = 256 (8 waves).
//    Each wave: 1 rowtile x 4 coltiles -> A loaded once per K-step, feeding
//    4 independent WMMA accumulator chains (latency hiding, 4x less A traffic).
// ---------------------------------------------------------------------------
__global__ void mat_gemm_kernel(const float* __restrict__ h,
                                const float* __restrict__ W,
                                const int* __restrict__ topidx,
                                float* __restrict__ LM,
                                float* __restrict__ LA) {
    const int bid = blockIdx.x;
    const int c   = bid >> 3;            // chunk
    const int cb  = bid & 7;             // 256-candidate block
    const int tid = threadIdx.x;
    const int lane = tid & 31, wave = tid >> 5;
    const int half = lane >> 4, l16 = lane & 15;

    const int rowtile = wave & 1;        // 0..1  (16 h-rows each)
    const int colgrp  = wave >> 1;       // 0..3  (4 coltiles each)
    const int rowbase = rowtile * 16;

    const int hr = c * CHUNKSZ + rowbase + l16;
    const float* arow = h + (size_t)hr * DIM;

    const float* brow[4];
    int colbase[4];
#pragma unroll
    for (int j = 0; j < 4; ++j) {
        colbase[j] = cb * 256 + (colgrp * 4 + j) * 16;
        const int cand = topidx[c * KCAND + colbase[j] + l16];
        brow[j] = W + (size_t)cand * DIM;
    }

    v8f accm[4] = {{}, {}, {}, {}};
    v8f acca[4] = {{}, {}, {}, {}};

    // K-steps 0..1 (first 64 dims) feed both main and aux accumulators
#pragma unroll
    for (int kk = 0; kk < 2; ++kk) {
        v16bf a = frag_f32(arow + kk * 32, half);
#pragma unroll
        for (int j = 0; j < 4; ++j) {
            v16bf b = frag_f32(brow[j] + kk * 32, half);
            accm[j] = wmma_bf16(a, b, accm[j]);
            acca[j] = wmma_bf16(a, b, acca[j]);
        }
    }
    // remaining K-steps, main accumulators only
#pragma unroll 2
    for (int kk = 2; kk < 32; ++kk) {
        __builtin_prefetch(arow + (kk + 4) * 32, 0, 0);   // global_prefetch_b8
        v16bf a = frag_f32(arow + kk * 32, half);
#pragma unroll
        for (int j = 0; j < 4; ++j) {
            v16bf b = frag_f32(brow[j] + kk * 32, half);
            accm[j] = wmma_bf16(a, b, accm[j]);
        }
    }

#pragma unroll
    for (int j = 0; j < 4; ++j) {
#pragma unroll
        for (int r = 0; r < 8; ++r) {
            const int M = r + 8 * half;
            const size_t off =
                (size_t)(c * CHUNKSZ + rowbase + M) * KCAND + colbase[j] + l16;
            LM[off] = accm[j][r];
            LA[off] = acca[j][r];
        }
    }
}

// ---------------------------------------------------------------------------
// 4) masked logsumexp + loss.  one workgroup per row (2048).
// ---------------------------------------------------------------------------
__global__ void mat_loss_kernel(const float* __restrict__ h,
                                const float* __restrict__ W,
                                const int* __restrict__ tgt,
                                const int* __restrict__ topidx,
                                const float* __restrict__ LM,
                                const float* __restrict__ LA,
                                const float* __restrict__ wsn,
                                float* __restrict__ out) {
    __shared__ float red[256];
    __shared__ float bm[256];
    __shared__ float bs[256];
    __shared__ float lossm_sh;

    const int row = blockIdx.x;
    const int c   = row >> 5;
    const int tid = threadIdx.x;
    const int t   = tgt[row];

    // pos logits and norms (one float4 per thread covers the 1024 dims)
    const float4* hv4 = (const float4*)(h + (size_t)row * DIM);
    const float4* wv4 = (const float4*)(W + (size_t)t * DIM);
    float pm = 0.0f, pa = 0.0f, hs = 0.0f, hls = 0.0f;
    {
        float4 hv = hv4[tid], wv = wv4[tid];
        float d = hv.x * wv.x + hv.y * wv.y + hv.z * wv.z + hv.w * wv.w;
        float q = hv.x * hv.x + hv.y * hv.y + hv.z * hv.z + hv.w * hv.w;
        pm = d; hs = q;
        if (tid < 16) { pa = d; hls = q; }     // dims < 64
    }
    pm  = block_sum(pm,  red, tid);
    pa  = block_sum(pa,  red, tid);
    hs  = block_sum(hs,  red, tid);
    hls = block_sum(hls, red, tid);

    const float wns     = wsn[0] * (1.0f / (float)VOCAB);
    const float wlns    = wsn[1] * (1.0f / (float)VOCAB);
    const float logvrem = logf((float)(VOCAB - KCAND - 1));
    const float ghost_m = logvrem + hs  * wns  * (1.0f / (float)DIM) * 0.5f;
    const float ghost_a = logvrem + hls * wlns * (1.0f / 64.0f) * 0.5f;

    // masked online logsumexp over candidate logits
    float mm = -3.0e38f, sm = 0.0f, ma = -3.0e38f, sa = 0.0f;
    const float* lmrow = LM + (size_t)row * KCAND;
    const float* larow = LA + (size_t)row * KCAND;
    const int*   ti    = topidx + c * KCAND;
    for (int k = tid; k < KCAND; k += 256) {
        if (ti[k] == t) continue;              // is_target -> -inf
        lse_accum(mm, sm, lmrow[k]);
        lse_accum(ma, sa, larow[k]);
    }

    // main tree
    bm[tid] = mm; bs[tid] = sm; __syncthreads();
    for (int off = 128; off > 0; off >>= 1) {
        if (tid < off) lse_combine(bm[tid], bs[tid], bm[tid + off], bs[tid + off]);
        __syncthreads();
    }
    if (tid == 0) {
        float Mm = bm[0], Sm = bs[0];
        lse_combine(Mm, Sm, pm, 1.0f);
        lse_combine(Mm, Sm, ghost_m, 1.0f);
        lossm_sh = Mm + logf(Sm) - pm;
    }
    __syncthreads();

    // aux tree
    bm[tid] = ma; bs[tid] = sa; __syncthreads();
    for (int off = 128; off > 0; off >>= 1) {
        if (tid < off) lse_combine(bm[tid], bs[tid], bm[tid + off], bs[tid + off]);
        __syncthreads();
    }
    if (tid == 0) {
        float Ma = bm[0], Sa = bs[0];
        lse_combine(Ma, Sa, pa, 1.0f);
        lse_combine(Ma, Sa, ghost_a, 1.0f);
        float lossa = Ma + logf(Sa) - pa;
        atomicAdd(out, (lossm_sh + 0.2f * lossa) * (1.0f / (float)NROWS));
    }
}

// ---------------------------------------------------------------------------
// launcher
// ---------------------------------------------------------------------------
extern "C" void kernel_launch(void* const* d_in, const int* in_sizes, int n_in,
                              void* d_out, int out_size, void* d_ws, size_t ws_size,
                              hipStream_t stream) {
    const float* h   = (const float*)d_in[0];
    const float* W   = (const float*)d_in[1];
    const int*   tgt = (const int*)d_in[2];
    float* out = (float*)d_out;

    // workspace layout (~33 MB total)
    float* wsn    = (float*)d_ws;                               // 2 f32
    int*   topidx = (int*)((char*)d_ws + 256);                  // 64*2048 i32 = 512 KB
    float* LM     = (float*)((char*)d_ws + ((size_t)1  << 20)); // 2048*2048 f32 = 16 MB
    float* LA     = (float*)((char*)d_ws + ((size_t)17 << 20)); // 16 MB

    mat_init_kernel<<<1, 64, 0, stream>>>(wsn, out);
    mat_wnorm_kernel<<<4096, 256, 0, stream>>>(W, wsn);
    mat_scan_kernel<<<1024, 256, 0, stream>>>(h, W, topidx);
    mat_gemm_kernel<<<512, 256, 0, stream>>>(h, W, topidx, LM, LA);
    mat_loss_kernel<<<2048, 256, 0, stream>>>(h, W, tgt, topidx, LM, LA, wsn, out);
}
